// VAEAttentionBlock_86895778332903
// MI455X (gfx1250) — matrix operand order repaired and checked by reading, not compiled, not run on hardware
//
#include <hip/hip_runtime.h>
#include <hip/hip_bf16.h>

#define DEVI __device__ __forceinline__

typedef __attribute__((ext_vector_type(16))) _Float16 v16h;
typedef __attribute__((ext_vector_type(8)))  _Float16 v8h;
typedef __attribute__((ext_vector_type(8)))  float    v8f;

constexpr int CH = 512;       // channels
constexpr int HW = 4096;      // spatial positions (64*64)
constexpr int BATCH = 4;
constexpr int GROUPS_N = 32;

constexpr size_t WELEM   = (size_t)CH * CH;          // 262144 halves per weight matrix
constexpr size_t OFF_XNT = 4 * WELEM;                // after wq,wk,wv,wo (f16)
constexpr size_t SZ_BNC  = (size_t)BATCH * HW * CH;  // 8388608 halves
constexpr size_t OFF_Q   = OFF_XNT + SZ_BNC;
constexpr size_t OFF_K   = OFF_Q + SZ_BNC;
constexpr size_t OFF_V   = OFF_K + SZ_BNC;
constexpr size_t OFF_O2  = OFF_V + SZ_BNC;

DEVI v8f vzero8() {
    v8f z;
#pragma unroll
    for (int i = 0; i < 8; i++) z[i] = 0.0f;
    return z;
}

DEVI v16h make16(v8h lo, v8h hi) {
    v16h r;
#pragma unroll
    for (int i = 0; i < 8; i++) { r[i] = lo[i]; r[8 + i] = hi[i]; }
    return r;
}

// A-matrix fragment (16x32 f16), A stored (M,K) row-major, K contiguous.
// lane m = lane%16; halves 0..7 -> K = k0 + 8*(lane/16) + 0..7; halves 8..15 -> +16.
DEVI v16h load_a_g(const _Float16* __restrict__ A, int lda, int m0, int k0, int lane) {
    const _Float16* p = A + (size_t)(m0 + (lane & 15)) * lda + k0 + ((lane >> 4) << 3);
    v8h lo = *(const v8h*)(p);
    v8h hi = *(const v8h*)(p + 16);
    return make16(lo, hi);
}

// B-matrix fragment (32x16 f16), B^T stored (N,K) row-major, K contiguous.
// lane n = lane%16; halves 0..7 -> K = k0 + 8*(lane/16) + 0..7; halves 8..15 -> +16.
DEVI v16h load_b_g(const _Float16* __restrict__ Bt, int ldb, int n0, int k0, int lane) {
    const _Float16* p = Bt + (size_t)(n0 + (lane & 15)) * ldb + k0 + ((lane >> 4) << 3);
    v8h lo = *(const v8h*)(p);
    v8h hi = *(const v8h*)(p + 16);
    return make16(lo, hi);
}

DEVI v8f wmma_f16(v16h a, v16h b, v8f c) {
    return __builtin_amdgcn_wmma_f32_16x16x32_f16(false, a, false, b, (short)0, c, false, false);
}

// ---------------- weight fp32 -> f16 ----------------
__global__ void __launch_bounds__(256)
wconv_kernel(const float* __restrict__ wq, const float* __restrict__ wk,
             const float* __restrict__ wv, const float* __restrict__ wo,
             _Float16* __restrict__ dst) {
    size_t i = (size_t)blockIdx.x * blockDim.x + threadIdx.x;
    if (i < WELEM) {
        dst[i]             = (_Float16)wq[i];
        dst[WELEM + i]     = (_Float16)wk[i];
        dst[2 * WELEM + i] = (_Float16)wv[i];
        dst[3 * WELEM + i] = (_Float16)wo[i];
    }
}

// ---------------- GroupNorm -> xnt (b, n, c) f16 ----------------
__global__ void __launch_bounds__(256)
gnorm_kernel(const float* __restrict__ x, const float* __restrict__ gamma,
             const float* __restrict__ beta, _Float16* __restrict__ xnt) {
    const int tid = threadIdx.x;
    const int b = blockIdx.x / GROUPS_N;
    const int g = blockIdx.x % GROUPS_N;
    const int CPG = CH / GROUPS_N;                 // 16 channels per group
    const float* base = x + (size_t)b * CH * HW + (size_t)g * CPG * HW;
    const int NE = CPG * HW;                        // 65536

    __shared__ float s1[256], s2[256];
    float sum = 0.f, ssq = 0.f;
    for (int i = tid; i < NE; i += 256) {
        float v = base[i];
        sum += v; ssq += v * v;
    }
    s1[tid] = sum; s2[tid] = ssq;
    __syncthreads();
    for (int s = 128; s > 0; s >>= 1) {
        if (tid < s) { s1[tid] += s1[tid + s]; s2[tid] += s2[tid + s]; }
        __syncthreads();
    }
    float mean = s1[0] * (1.0f / NE);
    float var  = s2[0] * (1.0f / NE) - mean * mean;
    float rstd = rsqrtf(var + 1e-5f);

    _Float16* out = xnt + (size_t)b * HW * CH;
    for (int i = tid; i < NE; i += 256) {
        int cc = i >> 12;               // channel within group (i / 4096)
        int n  = i & (HW - 1);
        int c  = g * CPG + cc;
        float v = (base[i] - mean) * rstd * gamma[c] + beta[c];
        out[(size_t)n * CH + c] = (_Float16)v;
    }
}

// ---------------- GEMM: D(m=spatial, n=cout) = Xnt * W^T; store (n,c) f16 ----------------
// outScale lets us fold the attention 1/sqrt(C) into the Q projection.
__global__ void __launch_bounds__(128)
proj_nc_kernel(const _Float16* __restrict__ Xnt, const _Float16* __restrict__ Wh,
               const float* __restrict__ bias, _Float16* __restrict__ Out, float outScale) {
    const int lane = threadIdx.x & 31;
    const int wave = threadIdx.x >> 5;
    const int b  = blockIdx.z;
    const int m0 = blockIdx.x * 64 + wave * 16;  // spatial
    const int n0 = blockIdx.y * 64;              // c_out
    const _Float16* A = Xnt + (size_t)b * HW * CH;

    v8f acc[4];
#pragma unroll
    for (int t = 0; t < 4; t++) acc[t] = vzero8();

    for (int k = 0; k < CH; k += 32) {
        v16h a = load_a_g(A, CH, m0, k, lane);
#pragma unroll
        for (int t = 0; t < 4; t++) {
            v16h bb = load_b_g(Wh, CH, n0 + 16 * t, k, lane);
            acc[t] = wmma_f16(a, bb, acc[t]);
        }
    }
    _Float16* O = Out + (size_t)b * HW * CH;
#pragma unroll
    for (int t = 0; t < 4; t++) {
        int col = n0 + 16 * t + (lane & 15);
        float bv = bias[col];
#pragma unroll
        for (int r = 0; r < 8; r++) {
            int row = m0 + r + ((lane >> 4) << 3);
            O[(size_t)row * CH + col] = (_Float16)((acc[t][r] + bv) * outScale);
        }
    }
}

// ---------------- GEMM: D(m=cout, n=spatial) = W * Xnt^T; store (c,n) f16 ----------------
__global__ void __launch_bounds__(128)
proj_cn_kernel(const _Float16* __restrict__ Wh, const _Float16* __restrict__ Xnt,
               const float* __restrict__ bias, _Float16* __restrict__ Out) {
    const int lane = threadIdx.x & 31;
    const int wave = threadIdx.x >> 5;
    const int b  = blockIdx.z;
    const int m0 = blockIdx.x * 64 + wave * 16;  // c_out
    const int n0 = blockIdx.y * 64;              // spatial
    const _Float16* Bt = Xnt + (size_t)b * HW * CH;

    v8f acc[4];
#pragma unroll
    for (int t = 0; t < 4; t++) acc[t] = vzero8();

    for (int k = 0; k < CH; k += 32) {
        v16h a = load_a_g(Wh, CH, m0, k, lane);
#pragma unroll
        for (int t = 0; t < 4; t++) {
            v16h bb = load_b_g(Bt, CH, n0 + 16 * t, k, lane);
            acc[t] = wmma_f16(a, bb, acc[t]);
        }
    }
    _Float16* O = Out + (size_t)b * CH * HW;
#pragma unroll
    for (int r = 0; r < 8; r++) {
        int row = m0 + r + ((lane >> 4) << 3);
        float bv = bias[row];
#pragma unroll
        for (int t = 0; t < 4; t++) {
            int col = n0 + 16 * t + (lane & 15);
            O[(size_t)row * HW + col] = (_Float16)(acc[t][r] + bv);
        }
    }
}

// ---------------- Flash attention, 16 queries per wave, single-pass online softmax ----------------
// Qt is pre-scaled by 1/sqrt(C), so logits = S directly.
__global__ void __launch_bounds__(128)
attn_kernel(const _Float16* __restrict__ Qt, const _Float16* __restrict__ Kt,
            const _Float16* __restrict__ Vf, _Float16* __restrict__ O2t) {
    const int lane = threadIdx.x & 31;
    const int wave = threadIdx.x >> 5;
    const int b  = blockIdx.y;
    const int i0 = blockIdx.x * 64 + wave * 16;     // query tile base
    const size_t nb = (size_t)b * HW * CH;
    const _Float16* Q  = Qt + nb;
    const _Float16* Kb = Kt + nb;
    const _Float16* Vb = Vf + nb;                   // V stored (c, n)

    __shared__ __align__(16) _Float16 pbuf[4][640]; // per-wave 16x40 P tile

    float rmax[8], rsum[8];
#pragma unroll
    for (int r = 0; r < 8; r++) { rmax[r] = -3.0e38f; rsum[r] = 0.f; }
    v8f o[32];
#pragma unroll
    for (int cb = 0; cb < 32; cb++) o[cb] = vzero8();

    for (int j0 = 0; j0 < HW; j0 += 32) {
        // ---- scores for this 32-key block ----
        v8f sA = vzero8(), sB = vzero8();
#pragma unroll 4
        for (int k = 0; k < CH; k += 32) {
            v16h a  = load_a_g(Q,  CH, i0, k, lane);
            v16h b0 = load_b_g(Kb, CH, j0,      k, lane);
            v16h b1 = load_b_g(Kb, CH, j0 + 16, k, lane);
            sA = wmma_f16(a, b0, sA);
            sB = wmma_f16(a, b1, sB);
        }
        // ---- online softmax update ----
        float f[8];
#pragma unroll
        for (int r = 0; r < 8; r++) {
            float bm = fmaxf(sA[r], sB[r]);
#pragma unroll
            for (int m = 1; m < 16; m <<= 1)
                bm = fmaxf(bm, __shfl_xor(bm, m));       // row max within half-wave
            float nm = fmaxf(rmax[r], bm);
            f[r] = __expf(rmax[r] - nm);                  // rescale factor (uniform per row)
            rmax[r] = nm;
        }
#pragma unroll
        for (int r = 0; r < 8; r++) {
            float p0 = __expf(sA[r] - rmax[r]);
            float p1 = __expf(sB[r] - rmax[r]);
            rsum[r] = rsum[r] * f[r] + p0 + p1;
            int row = r + ((lane >> 4) << 3);
            pbuf[wave][row * 40 + (lane & 15)]      = (_Float16)p0;
            pbuf[wave][row * 40 + 16 + (lane & 15)] = (_Float16)p1;
        }
        __syncthreads();
        // reload P as A-fragment (16x32) from LDS
        v16h pa;
        {
            const _Float16* p = &pbuf[wave][(lane & 15) * 40 + ((lane >> 4) << 3)];
            v8h lo = *(const v8h*)(p);
            v8h hi = *(const v8h*)(p + 16);
            pa = make16(lo, hi);
        }
        // ---- rescale O and accumulate P*V (rescale VALU co-executes with XDL WMMA) ----
#pragma unroll
        for (int cb = 0; cb < 32; cb++) {
            v8f t = o[cb];
#pragma unroll
            for (int r = 0; r < 8; r++) t[r] *= f[r];
            v16h vb = load_b_g(Vb, HW, cb * 16, j0, lane);
            o[cb] = wmma_f16(pa, vb, t);
        }
        __syncthreads();
    }
#pragma unroll
    for (int r = 0; r < 8; r++) {
#pragma unroll
        for (int m = 1; m < 16; m <<= 1)
            rsum[r] += __shfl_xor(rsum[r], m);
    }

    _Float16* Ob = O2t + nb;                        // (n, c) layout
#pragma unroll
    for (int cb = 0; cb < 32; cb++) {
#pragma unroll
        for (int r = 0; r < 8; r++) {
            int row = i0 + r + ((lane >> 4) << 3);
            float val = o[cb][r] / rsum[r];
            Ob[(size_t)row * CH + cb * 16 + (lane & 15)] = (_Float16)val;
        }
    }
}

// ---------------- final: out = x + Wo*out2 + bo (f32) ----------------
__global__ void __launch_bounds__(128)
final_kernel(const _Float16* __restrict__ Woh, const _Float16* __restrict__ O2t,
             const float* __restrict__ bo, const float* __restrict__ x,
             float* __restrict__ out) {
    const int lane = threadIdx.x & 31;
    const int wave = threadIdx.x >> 5;
    const int b  = blockIdx.z;
    const int m0 = blockIdx.x * 64 + wave * 16;  // c_out
    const int n0 = blockIdx.y * 64;              // spatial
    const _Float16* Bt = O2t + (size_t)b * HW * CH;

    v8f acc[4];
#pragma unroll
    for (int t = 0; t < 4; t++) acc[t] = vzero8();

    for (int k = 0; k < CH; k += 32) {
        v16h a = load_a_g(Woh, CH, m0, k, lane);
#pragma unroll
        for (int t = 0; t < 4; t++) {
            v16h bb = load_b_g(Bt, CH, n0 + 16 * t, k, lane);
            acc[t] = wmma_f16(a, bb, acc[t]);
        }
    }
#pragma unroll
    for (int r = 0; r < 8; r++) {
        int row = m0 + r + ((lane >> 4) << 3);
        float bv = bo[row];
        size_t rb = ((size_t)b * CH + row) * HW;
#pragma unroll
        for (int t = 0; t < 4; t++) {
            int col = n0 + 16 * t + (lane & 15);
            out[rb + col] = acc[t][r] + bv + x[rb + col];
        }
    }
}

extern "C" void kernel_launch(void* const* d_in, const int* in_sizes, int n_in,
                              void* d_out, int out_size, void* d_ws, size_t ws_size,
                              hipStream_t stream) {
    (void)in_sizes; (void)n_in; (void)out_size; (void)ws_size;
    const float* x     = (const float*)d_in[0];
    const float* gamma = (const float*)d_in[1];
    const float* beta  = (const float*)d_in[2];
    const float* wq    = (const float*)d_in[3];
    const float* bq    = (const float*)d_in[4];
    const float* wk    = (const float*)d_in[5];
    const float* bk    = (const float*)d_in[6];
    const float* wv    = (const float*)d_in[7];
    const float* bv    = (const float*)d_in[8];
    const float* wo    = (const float*)d_in[9];
    const float* bo    = (const float*)d_in[10];

    _Float16* ws0 = (_Float16*)d_ws;
    _Float16* xnt = ws0 + OFF_XNT;
    _Float16* Qt  = ws0 + OFF_Q;
    _Float16* Kt  = ws0 + OFF_K;
    _Float16* Vf  = ws0 + OFF_V;
    _Float16* O2t = ws0 + OFF_O2;

    const float qscale = 0.044194173824159216f;   // 1/sqrt(512)

    wconv_kernel<<<dim3((unsigned)((WELEM + 255) / 256)), 256, 0, stream>>>(wq, wk, wv, wo, ws0);
    gnorm_kernel<<<dim3(BATCH * GROUPS_N), 256, 0, stream>>>(x, gamma, beta, xnt);
    proj_nc_kernel<<<dim3(64, 8, BATCH), 128, 0, stream>>>(xnt, ws0,         bq, Qt, qscale);
    proj_nc_kernel<<<dim3(64, 8, BATCH), 128, 0, stream>>>(xnt, ws0 + WELEM, bk, Kt, 1.0f);
    proj_cn_kernel<<<dim3(8, 64, BATCH), 128, 0, stream>>>(ws0 + 2 * WELEM, xnt, bv, Vf);
    attn_kernel<<<dim3(64, BATCH), 128, 0, stream>>>(Qt, Kt, Vf, O2t);
    final_kernel<<<dim3(8, 64, BATCH), 128, 0, stream>>>(ws0 + 3 * WELEM, O2t, bo, x, (float*)d_out);
}
